// NODENet_72653666779173
// MI455X (gfx1250) — matrix operand order repaired and checked by reading, not compiled, not run on hardware
//
#include <hip/hip_runtime.h>
#include <hip/hip_bf16.h>
#include <math.h>

typedef __attribute__((ext_vector_type(16))) __bf16 v16bf;
typedef __attribute__((ext_vector_type(8)))  __bf16 v8bf;
typedef __attribute__((ext_vector_type(8)))  float  v8f;

union bf16x16 { v16bf v; v8bf h[2]; };

#define B_DIM 8192
#define T_DIM 128
#define D_DIM 6
#define L_DIM 64
#define N_DIM 768   // T*D
#define F_DIM 256

// ---------------------------------------------------------------------------
// f32 -> bf16 elementwise convert
// ---------------------------------------------------------------------------
__global__ void convert_bf16_kernel(const float* __restrict__ in,
                                    __bf16* __restrict__ out, int n) {
    int i = blockIdx.x * blockDim.x + threadIdx.x;
    if (i < n) out[i] = (__bf16)in[i];
}

// ---------------------------------------------------------------------------
// row softmax (one wave per row), emit bf16 weights in natural (N,K) layout
// ---------------------------------------------------------------------------
__global__ void softmax_rows_bf16_kernel(const float* __restrict__ in,
                                         __bf16* __restrict__ out, int cols) {
    const int row  = blockIdx.x;
    const int lane = threadIdx.x;            // blockDim.x == 32 (wave32)
    const float* p = in  + (size_t)row * cols;
    __bf16*      q = out + (size_t)row * cols;

    float m = -INFINITY;
    for (int c = lane; c < cols; c += 32) m = fmaxf(m, p[c]);
    #pragma unroll
    for (int o = 16; o > 0; o >>= 1) m = fmaxf(m, __shfl_xor(m, o, 32));

    float s = 0.0f;
    for (int c = lane; c < cols; c += 32) s += __expf(p[c] - m);
    #pragma unroll
    for (int o = 16; o > 0; o >>= 1) s += __shfl_xor(s, o, 32);

    const float inv = 1.0f / s;
    for (int c = lane; c < cols; c += 32)
        q[c] = (__bf16)(__expf(p[c] - m) * inv);
}

// ---------------------------------------------------------------------------
// WMMA GEMM: C(MxN f32, row-major) = A(MxK bf16, row-major) * B^T
//            B stored as (N x K) bf16 row-major (natural fw[n][k] layout).
// Each wave computes a 16(M) x 64(N) strip: A fragment loaded once per
// K-step, reused across 4 WMMAs. 4 waves (4 M-tiles) per block.
// ---------------------------------------------------------------------------
__global__ void wmma_gemm_bf16_kernel(const __bf16* __restrict__ A,
                                      const __bf16* __restrict__ Bt,
                                      float* __restrict__ C,
                                      int M, int N, int K) {
    const int lane = threadIdx.x & 31;
    const int wave = threadIdx.x >> 5;
    const int row  = lane & 15;
    const int half = lane >> 4;

    const int n0 = blockIdx.x * 64;
    const int m0 = (blockIdx.y * 4 + wave) * 16;
    if (m0 >= M || n0 >= N) return;

    const __bf16* aptr = A  + (size_t)(m0 + row) * K;
    const __bf16* bptr = Bt + (size_t)(n0 + row) * K;   // tile t adds t*16*K

    v8f c0 = {}, c1 = {}, c2 = {}, c3 = {};
    for (int k0 = 0; k0 < K; k0 += 32) {
        // prefetch next K block (no-op past the end: speculative prefetch)
        __builtin_prefetch(aptr + k0 + 32, 0, 3);
        __builtin_prefetch(bptr + k0 + 32, 0, 3);

        // A frag: lanes 0-15 -> K {0..15}; lanes 16-31 -> K {16..31}
        bf16x16 a;
        a.h[0] = *(const v8bf*)(aptr + k0 + half * 16);
        a.h[1] = *(const v8bf*)(aptr + k0 + half * 16 + 8);

        // B frags: lanes 0-15 -> K {0..15}; lanes 16-31 -> K {16..31}; col = lane&15
        bf16x16 b0, b1, b2, b3;
        b0.h[0] = *(const v8bf*)(bptr + k0 + half * 16);
        b0.h[1] = *(const v8bf*)(bptr + k0 + half * 16 + 8);
        b1.h[0] = *(const v8bf*)(bptr + (size_t)16 * K + k0 + half * 16);
        b1.h[1] = *(const v8bf*)(bptr + (size_t)16 * K + k0 + half * 16 + 8);
        b2.h[0] = *(const v8bf*)(bptr + (size_t)32 * K + k0 + half * 16);
        b2.h[1] = *(const v8bf*)(bptr + (size_t)32 * K + k0 + half * 16 + 8);
        b3.h[0] = *(const v8bf*)(bptr + (size_t)48 * K + k0 + half * 16);
        b3.h[1] = *(const v8bf*)(bptr + (size_t)48 * K + k0 + half * 16 + 8);

        c0 = __builtin_amdgcn_wmma_f32_16x16x32_bf16(false, a.v, false, b0.v,
                                                     (short)0, c0, false, false);
        c1 = __builtin_amdgcn_wmma_f32_16x16x32_bf16(false, a.v, false, b1.v,
                                                     (short)0, c1, false, false);
        c2 = __builtin_amdgcn_wmma_f32_16x16x32_bf16(false, a.v, false, b2.v,
                                                     (short)0, c2, false, false);
        c3 = __builtin_amdgcn_wmma_f32_16x16x32_bf16(false, a.v, false, b3.v,
                                                     (short)0, c3, false, false);
    }

    // C/D layout: VGPR r -> row M = half*8 + r, col N = lane&15
    float* cptr = C + (size_t)(m0 + half * 8) * N + n0 + (lane & 15);
    #pragma unroll
    for (int r = 0; r < 8; ++r) {
        cptr[(size_t)r * N + 0]  = c0[r];
        cptr[(size_t)r * N + 16] = c1[r];
        cptr[(size_t)r * N + 32] = c2[r];
        cptr[(size_t)r * N + 48] = c3[r];
    }
}

// ---------------------------------------------------------------------------
// Oblivious-tree evaluation: one thread per (b, t).
// sel row is contiguous (t*6 + d); 64-leaf probs built by in-register doubling.
// ---------------------------------------------------------------------------
__global__ void tree_kernel(const float* __restrict__ sel,
                            const float* __restrict__ th,
                            const float* __restrict__ lt,
                            const float* __restrict__ lr,
                            float* __restrict__ h) {
    int idx = blockIdx.x * blockDim.x + threadIdx.x;
    if (idx >= B_DIM * T_DIM) return;
    const int b = idx >> 7;
    const int t = idx & (T_DIM - 1);

    const float* sp = sel + (size_t)b * N_DIM + t * D_DIM;
    float dec[D_DIM];
    #pragma unroll
    for (int d = 0; d < D_DIM; ++d) {
        const float z = (sp[d] - th[t * D_DIM + d]) * __expf(lt[t * D_DIM + d]);
        dec[d] = 1.0f / (1.0f + __expf(-z));
    }

    // leaf index bits: depth d contributes bit (5-d); bit==1 -> dec[d]
    float p[L_DIM];
    p[0] = 1.0f;
    #pragma unroll
    for (int d = 0; d < D_DIM; ++d) {
        const int sz = 1 << d;
        #pragma unroll
        for (int i = sz - 1; i >= 0; --i) {
            const float v = p[i];
            p[2 * i + 1] = v * dec[d];
            p[2 * i]     = v * (1.0f - dec[d]);
        }
    }

    const float* lp = lr + (size_t)t * L_DIM;
    float out = 0.0f;
    #pragma unroll
    for (int l = 0; l < L_DIM; ++l) out += p[l] * lp[l];
    h[(size_t)b * T_DIM + t] = out;
}

// ---------------------------------------------------------------------------
// BN training stats: one block per column, reduce over 8192 rows
// ---------------------------------------------------------------------------
__global__ void bn_stats_kernel(const float* __restrict__ h,
                                float* __restrict__ mu,
                                float* __restrict__ var) {
    const int c   = blockIdx.x;       // 0..127
    const int tid = threadIdx.x;      // blockDim.x == 256
    float s = 0.0f, s2 = 0.0f;
    for (int b = tid; b < B_DIM; b += 256) {
        const float v = h[(size_t)b * T_DIM + c];
        s += v; s2 += v * v;
    }
    __shared__ float rs[256], rs2[256];
    rs[tid] = s; rs2[tid] = s2;
    __syncthreads();
    for (int o = 128; o > 0; o >>= 1) {
        if (tid < o) { rs[tid] += rs[tid + o]; rs2[tid] += rs2[tid + o]; }
        __syncthreads();
    }
    if (tid == 0) {
        const float m = rs[0] * (1.0f / B_DIM);
        mu[c]  = m;
        var[c] = rs2[0] * (1.0f / B_DIM) - m * m;   // biased var, matches jnp.var
    }
}

// ---------------------------------------------------------------------------
// BN apply + ReLU; emit bf16 (next GEMM's A operand) and optionally f32
// ---------------------------------------------------------------------------
__global__ void bn_apply_relu_kernel(const float* __restrict__ h,
                                     const float* __restrict__ mu,
                                     const float* __restrict__ var,
                                     const float* __restrict__ g,
                                     const float* __restrict__ be,
                                     __bf16* __restrict__ out_bf,
                                     float* __restrict__ out_f) {
    int i = blockIdx.x * blockDim.x + threadIdx.x;
    if (i >= B_DIM * T_DIM) return;
    const int c = i & (T_DIM - 1);
    float y = g[c] * (h[i] - mu[c]) * rsqrtf(var[c] + 1e-5f) + be[c];
    y = fmaxf(y, 0.0f);
    out_bf[i] = (__bf16)y;
    if (out_f) out_f[i] = y;
}

// ---------------------------------------------------------------------------
// Head MLP: relu(h@w1+b1)@w2+b2 ; one thread per sample, w1 staged in LDS
// ---------------------------------------------------------------------------
__global__ void mlp_kernel(const float* __restrict__ h,
                           const float* __restrict__ w1,
                           const float* __restrict__ b1,
                           const float* __restrict__ w2,
                           const float* __restrict__ b2,
                           float* __restrict__ out) {
    __shared__ float sw1[T_DIM * 64];
    for (int i = threadIdx.x; i < T_DIM * 64; i += blockDim.x) sw1[i] = w1[i];
    __syncthreads();

    const int b = blockIdx.x * blockDim.x + threadIdx.x;
    if (b >= B_DIM) return;
    const float* row = h + (size_t)b * T_DIM;

    float acc = b2[0];
    #pragma unroll 4
    for (int j = 0; j < 64; ++j) {
        float s = b1[j];
        for (int k = 0; k < T_DIM; ++k) s += row[k] * sw1[k * 64 + j];
        acc += fmaxf(s, 0.0f) * w2[j];
    }
    out[b] = acc;
}

// ---------------------------------------------------------------------------
extern "C" void kernel_launch(void* const* d_in, const int* in_sizes, int n_in,
                              void* d_out, int out_size, void* d_ws, size_t ws_size,
                              hipStream_t stream) {
    const float* x   = (const float*)d_in[0];
    const float* fs1 = (const float*)d_in[1];
    const float* th1 = (const float*)d_in[2];
    const float* lr1 = (const float*)d_in[3];
    const float* lt1 = (const float*)d_in[4];
    const float* g1  = (const float*)d_in[5];
    const float* be1 = (const float*)d_in[6];
    const float* fs2 = (const float*)d_in[7];
    const float* th2 = (const float*)d_in[8];
    const float* lr2 = (const float*)d_in[9];
    const float* lt2 = (const float*)d_in[10];
    const float* g2  = (const float*)d_in[11];
    const float* be2 = (const float*)d_in[12];
    const float* w1  = (const float*)d_in[13];
    const float* b1  = (const float*)d_in[14];
    const float* w2  = (const float*)d_in[15];
    const float* b2  = (const float*)d_in[16];
    float* out = (float*)d_out;

    // workspace carve-up (all offsets 256B aligned)
    char* ws = (char*)d_ws;
    __bf16* xb1  = (__bf16*)(ws);                         // 8192*256*2 = 4 MB
    __bf16* fw1b = (__bf16*)(ws + 4194304);               // 768*256*2
    __bf16* fw2b = (__bf16*)(ws + 4587520);               // 768*128*2
    float*  sel  = (float*)(ws + 4784128);                // 8192*768*4 = 24 MB
    float*  h    = (float*)(ws + 29949952);               // 8192*128*4 = 4 MB
    __bf16* xb2  = (__bf16*)(ws + 34144256);              // 8192*128*2 = 2 MB
    float*  h2n  = (float*)(ws + 36241408);               // 8192*128*4 = 4 MB
    float*  mu   = (float*)(ws + 40435712);               // 128*4
    float*  var  = (float*)(ws + 40436224);               // 128*4

    // ---- block 1 ----
    convert_bf16_kernel<<<(B_DIM * F_DIM + 255) / 256, 256, 0, stream>>>(x, xb1, B_DIM * F_DIM);
    softmax_rows_bf16_kernel<<<N_DIM, 32, 0, stream>>>(fs1, fw1b, F_DIM);
    softmax_rows_bf16_kernel<<<N_DIM, 32, 0, stream>>>(fs2, fw2b, T_DIM);

    wmma_gemm_bf16_kernel<<<dim3(N_DIM / 64, B_DIM / 64), 128, 0, stream>>>(
        xb1, fw1b, sel, B_DIM, N_DIM, F_DIM);
    tree_kernel<<<(B_DIM * T_DIM + 255) / 256, 256, 0, stream>>>(sel, th1, lt1, lr1, h);
    bn_stats_kernel<<<T_DIM, 256, 0, stream>>>(h, mu, var);
    bn_apply_relu_kernel<<<(B_DIM * T_DIM + 255) / 256, 256, 0, stream>>>(
        h, mu, var, g1, be1, xb2, (float*)nullptr);

    // ---- block 2 ----
    wmma_gemm_bf16_kernel<<<dim3(N_DIM / 64, B_DIM / 64), 128, 0, stream>>>(
        xb2, fw2b, sel, B_DIM, N_DIM, T_DIM);
    tree_kernel<<<(B_DIM * T_DIM + 255) / 256, 256, 0, stream>>>(sel, th2, lt2, lr2, h);
    bn_stats_kernel<<<T_DIM, 256, 0, stream>>>(h, mu, var);
    bn_apply_relu_kernel<<<(B_DIM * T_DIM + 255) / 256, 256, 0, stream>>>(
        h, mu, var, g2, be2, xb2 /*dummy sink*/, h2n);

    // ---- head ----
    mlp_kernel<<<(B_DIM + 255) / 256, 256, 0, stream>>>(h2n, w1, b1, w2, b2, out);

    (void)in_sizes; (void)n_in; (void)out_size; (void)ws_size;
}
